// SSIM3D_30090540876078
// MI455X (gfx1250) — matrix operand order repaired and checked by reading, not compiled, not run on hardware
//
#include <hip/hip_runtime.h>

typedef _Float16 v16h __attribute__((ext_vector_type(16)));
typedef _Float16 v8h  __attribute__((ext_vector_type(8)));
typedef float    v8f  __attribute__((ext_vector_type(8)));

#define PITCH 24   // padded row pitch (f16 elems): 20 valid + 4 zero pad, 48B = 16B-aligned rows

// Staging array sizes (f16 elems), each with tail slack so the K=24..31 over-read
// of the last row stays in-array. Pads/slack are zeroed once and never written again;
// the W band matrix is zero for K>=20, so those lanes contribute exactly 0.
#define XT_SZ (400 * PITCH + 16)   // 9616
#define T1_SZ (320 * PITCH + 16)   // 7696
#define T2_SZ (256 * PITCH + 16)   // 6160

// One 16-row x 16-col 5-tap 1-D convolution slab:
//   D(16x16 f32) = A(16x32 f16) x bW(32x16 f16 band) + 0
// A layout per ISA 7.12.2 (16-bit A 16x32, wave32): lane l -> row m=l&15, half h=l>>4;
// elements 0..7 <-> K=8h..8h+7 (contiguous), elements 8..15 <-> K=16+8h..16+8h+7.
// With PITCH=24 both halves are 16B-aligned -> two ds_load_b128 per operand.
// MODE==0: A = p rows;  MODE==1: A = (p*q) rows (fields x^2, y^2, xy formed on the fly).
template <int MODE>
__device__ __forceinline__ v8f conv_rows16(const _Float16* __restrict__ p,
                                           const _Float16* __restrict__ q,
                                           int r0, v16h bW, int lane)
{
    const int m = lane & 15;
    const int h = lane >> 4;
    const int base = (r0 + m) * PITCH + 8 * h;   // byte addr = 48*(r0+m)+16h : 16B aligned

    v8h lo = *(const v8h*)(p + base);
    v8h hi = *(const v8h*)(p + base + 16);
    if (MODE == 1) {
        lo *= *(const v8h*)(q + base);
        hi *= *(const v8h*)(q + base + 16);
    }
    v16h a;
#pragma unroll
    for (int e = 0; e < 8; ++e) { a[e] = lo[e]; a[8 + e] = hi[e]; }

    v8f c = {};
    // (neg_a, A, neg_b, B, c_mod, C, reuse_a, reuse_b)
    return __builtin_amdgcn_wmma_f32_16x16x32_f16(false, a, false, bW,
                                                  (short)0, c, false, false);
}

__global__ __launch_bounds__(256)
void ssim3d_tiles(const float* __restrict__ xg, const float* __restrict__ yg,
                  const float* __restrict__ k3, float* __restrict__ partial)
{
    __shared__ _Float16 XT[XT_SZ];      // x-hat halo tile, rows (z*20+y), pitch 24
    __shared__ _Float16 YT[XT_SZ];      // y-hat halo tile
    __shared__ _Float16 T1[T1_SZ];      // after x-blur: rows (z*16+xo), elem y
    __shared__ _Float16 T2[T2_SZ];      // after y-blur: rows (yo*16+xo), elem z
    __shared__ float    OUTF[5 * 4096]; // 5 fully-blurred fields, f32
    __shared__ float    red[256];

    const int tid  = threadIdx.x;
    const int lane = tid & 31;
    const int wave = tid >> 5;

    // Zero all staging (row pads + tail slack must be finite-zero; see note above).
    for (int t = tid; t < XT_SZ; t += 256) { XT[t] = (_Float16)0.0f; YT[t] = (_Float16)0.0f; }
    for (int t = tid; t < T1_SZ; t += 256) T1[t] = (_Float16)0.0f;
    for (int t = tid; t < T2_SZ; t += 256) T2[t] = (_Float16)0.0f;

    // 1-D weights from the provided (4,1,5,5,5) kernel:
    // k3[0,0,2,2,t] = w2*w2*w[t], w2 = cbrt(k3[center])
    float w1d[5];
    {
        const float w2  = cbrtf(k3[62]);
        const float inv = 1.0f / (w2 * w2);
#pragma unroll
        for (int t = 0; t < 5; ++t) w1d[t] = k3[60 + t] * inv;
    }

    // Constant B operand (32x16 f16): W[k][n] = w1d[k-n] for 0<=k-n<=4, else 0.
    // B layout: lane l -> col n=l&15, half h=l>>4; element e<8 -> K = 8h+e, e>=8 -> K = 16+8h+(e-8).
    v16h bW;
    {
        const int n = lane & 15, h = lane >> 4;
#pragma unroll
        for (int e = 0; e < 16; ++e) {
            const int k  = (e < 8) ? (8 * h + e) : (16 + 8 * h + (e - 8));
            const int dk = k - n;
            bW[e] = (dk >= 0 && dk <= 4) ? (_Float16)w1d[dk] : (_Float16)0.0f;
        }
    }

    // Tile decode: 8 (n*c) x 8x8x8 tiles of 16^3
    const int b  = blockIdx.x;
    const int nc = b >> 9;
    const int t9 = b & 511;
    const int tz = t9 >> 6, ty = (t9 >> 3) & 7, tx = t9 & 7;
    const int z0 = tz * 16 - 2, y0 = ty * 16 - 2, x0 = tx * 16 - 2;
    const int base = nc * (128 * 128 * 128);

    __syncthreads();   // memset before halo-load writes

    // Load halo tiles; apply (v+1)*0.5; zero-pad outside the volume.
    // Logical t in [0,8000): row = t/20 (= z*20+y), col = t%20.
    for (int t = tid; t < 8000; t += 256) {
        const int row = t / 20;
        const int lx  = t - row * 20;
        const int lz  = row / 20;
        const int ly  = row - lz * 20;
        const int gz = z0 + lz, gy = y0 + ly, gx = x0 + lx;
        float xv = 0.0f, yv = 0.0f;
        if ((unsigned)gz < 128u && (unsigned)gy < 128u && (unsigned)gx < 128u) {
            const int gi = base + (gz << 14) + (gy << 7) + gx;
            xv = (xg[gi] + 1.0f) * 0.5f;
            yv = (yg[gi] + 1.0f) * 0.5f;
        }
        const int idx = row * PITCH + lx;
        XT[idx] = (_Float16)xv;
        YT[idx] = (_Float16)yv;
    }
    __syncthreads();

    // Five fields: mu1<-x, mu2<-y, E[x^2], E[y^2], E[xy]; each: 3 WMMA blur passes.
    for (int f = 0; f < 5; ++f) {
        const _Float16* p = (f == 1 || f == 3) ? YT : XT;
        const _Float16* q = (f == 3 || f == 4) ? YT : XT;   // used only when f>=2
        const bool prod = (f >= 2);

        // Pass X: rows r=(z*20+y) in [0,400), blur along x -> T1[(z*16+xo)*24 + y]
        for (int ch = wave; ch < 25; ch += 8) {
            const v8f d = prod ? conv_rows16<1>(p, q, ch * 16, bW, lane)
                               : conv_rows16<0>(p, p, ch * 16, bW, lane);
            const int n = lane & 15, h = lane >> 4;
            int r = ch * 16 + 8 * h;
            int z = r / 20;                     // one division per chunk
            int y = r - z * 20;
            int rowbase = (z * 16 + n) * PITCH; // T1 row (z, n)
#pragma unroll
            for (int j = 0; j < 8; ++j) {
                T1[rowbase + y] = (_Float16)d[j];
                if (++y == 20) { y = 0; rowbase += 16 * PITCH; }
            }
        }
        __syncthreads();

        // Pass Y: rows r=(z*16+xo) in [0,320), blur along y -> T2[(yo*16+xo)*24 + z]
        for (int ch = wave; ch < 20; ch += 8) {
            const v8f d = conv_rows16<0>(T1, T1, ch * 16, bW, lane);
            const int n = lane & 15, h = lane >> 4;
#pragma unroll
            for (int j = 0; j < 8; ++j) {
                const int r = ch * 16 + 8 * h + j;
                const int z = r >> 4, xo = r & 15;
                T2[(n * 16 + xo) * PITCH + z] = (_Float16)d[j];
            }
        }
        __syncthreads();

        // Pass Z: rows r=(yo*16+xo) in [0,256), blur along z -> OUTF[f][zo][yo][xo] (f32)
        for (int ch = wave; ch < 16; ch += 8) {
            const v8f d = conv_rows16<0>(T2, T2, ch * 16, bW, lane);
            const int n = lane & 15, h = lane >> 4;
#pragma unroll
            for (int j = 0; j < 8; ++j) {
                const int r  = ch * 16 + 8 * h + j;
                const int yo = r >> 4, xo = r & 15;
                OUTF[f * 4096 + n * 256 + yo * 16 + xo] = d[j];
            }
        }
        __syncthreads();  // T1/T2 reuse next field + OUTF visibility
    }

    // SSIM map (all f32) + deterministic block reduction
    const float c1 = 1.0e-4f, c2 = 9.0e-4f;
    float acc = 0.0f;
    for (int t = tid; t < 4096; t += 256) {
        const float m1  = OUTF[t];
        const float m2  = OUTF[4096 + t];
        const float sxx = OUTF[2 * 4096 + t];
        const float syy = OUTF[3 * 4096 + t];
        const float sxy = OUTF[4 * 4096 + t];
        const float m1s = m1 * m1, m2s = m2 * m2, m12 = m1 * m2;
        const float s1 = sxx - m1s, s2 = syy - m2s, s12 = sxy - m12;
        const float num = (2.0f * m12 + c1) * (2.0f * s12 + c2);
        const float den = (m1s + m2s + c1) * (s1 + s2 + c2);
        acc += num / den;
    }
    red[tid] = acc;
    __syncthreads();
#pragma unroll
    for (int s = 128; s > 0; s >>= 1) {
        if (tid < s) red[tid] += red[tid + s];
        __syncthreads();
    }
    if (tid == 0) partial[b] = red[0];
}

__global__ __launch_bounds__(256)
void ssim3d_reduce(const float* __restrict__ partial, float* __restrict__ out,
                   int n, float scale)
{
    __shared__ float red[256];
    float acc = 0.0f;
    for (int i = threadIdx.x; i < n; i += 256) acc += partial[i];
    red[threadIdx.x] = acc;
    __syncthreads();
#pragma unroll
    for (int s = 128; s > 0; s >>= 1) {
        if ((int)threadIdx.x < s) red[threadIdx.x] += red[threadIdx.x + s];
        __syncthreads();
    }
    if (threadIdx.x == 0) out[0] = red[0] * scale;
}

extern "C" void kernel_launch(void* const* d_in, const int* in_sizes, int n_in,
                              void* d_out, int out_size, void* d_ws, size_t ws_size,
                              hipStream_t stream) {
    (void)in_sizes; (void)n_in; (void)out_size; (void)ws_size;
    const float* x  = (const float*)d_in[0];
    const float* y  = (const float*)d_in[1];
    const float* k3 = (const float*)d_in[2];
    float* ws  = (float*)d_ws;   // 4096 block partials (16 KB)
    float* out = (float*)d_out;  // single f32 scalar

    ssim3d_tiles<<<dim3(4096), dim3(256), 0, stream>>>(x, y, k3, ws);
    ssim3d_reduce<<<dim3(1), dim3(256), 0, stream>>>(ws, out, 4096,
                                                     1.0f / 16777216.0f);
}
